// BertAttention_55336358642449
// MI455X (gfx1250) — matrix operand order repaired and checked by reading, not compiled, not run on hardware
//
#include <hip/hip_runtime.h>
#include <math.h>

typedef __attribute__((ext_vector_type(16))) _Float16 v16h;
typedef __attribute__((ext_vector_type(8)))  _Float16 v8h;
typedef __attribute__((ext_vector_type(8)))  float    v8f;

static constexpr int Bb = 2, S = 2048, D = 1024, H = 16, HD = 64;
static constexpr int M  = Bb * S;      // 4096 rows

// ---------- helpers ----------
__device__ __forceinline__ v8h ld8(const _Float16* p) { return *(const v8h*)p; }

__device__ __forceinline__ v16h cat16(v8h a, v8h b) {
    return __builtin_shufflevector(a, b, 0,1,2,3,4,5,6,7,8,9,10,11,12,13,14,15);
}

__device__ __forceinline__ v8f wmma16(v16h a, v16h b, v8f c) {
    // v_wmma_f32_16x16x32_f16 : D = A(16x32 f16) * B(32x16 f16) + C(16x16 f32)
    return __builtin_amdgcn_wmma_f32_16x16x32_f16(false, a, false, b, (short)0, c, false, false);
}

__device__ __forceinline__ v8f zero8() {
    v8f v;
#pragma unroll
    for (int i = 0; i < 8; ++i) v[i] = 0.f;
    return v;
}

// exchange with lane^16 partner (wave32) via LDS-permute hardware (no LDS memory)
__device__ __forceinline__ float xor16f(float x) {
    int lane = (int)(threadIdx.x & 31u);
    int addr = ((lane ^ 16) & 31) << 2;
    return __int_as_float(__builtin_amdgcn_ds_bpermute(addr, __float_as_int(x)));
}

// ---------- prep kernels ----------
__global__ void k_cvt_f16(const float* __restrict__ x, _Float16* __restrict__ y, int n) {
    int i = blockIdx.x * blockDim.x + threadIdx.x;
    if (i < n) y[i] = (_Float16)x[i];
}

// wt[n][k] = (f16) w[k][n]   (1024x1024)
__global__ void k_transpose_f16(const float* __restrict__ w, _Float16* __restrict__ wt) {
    int i = blockIdx.x * blockDim.x + threadIdx.x;   // over 2^20 elements
    int n = i >> 10, k = i & 1023;
    wt[i] = (_Float16)w[(size_t)k * 1024 + n];
}

// ---------- fused QKV GEMM ----------
// grid (M/32, D/64, 3); block 32 (one wave). Each wave: 32 rows x 64 cols strip
// (2 A-frags, each B-frag reused by both -> 8 WMMAs per 192B/lane of L2 traffic).
__global__ void k_qkv_gemm(const _Float16* __restrict__ X,
                           const _Float16* __restrict__ WtQ,
                           const _Float16* __restrict__ WtK,
                           const _Float16* __restrict__ WtV,
                           const float* __restrict__ bq,
                           const float* __restrict__ bk,
                           const float* __restrict__ bv,
                           _Float16* __restrict__ Q,
                           _Float16* __restrict__ K,
                           _Float16* __restrict__ Vt) {
    const int lane = threadIdx.x & 31;
    const int n = lane & 15, half = lane >> 4;
    const int mbase = blockIdx.x * 32;
    const int nbase = blockIdx.y * 64;
    const int z = blockIdx.z;
    const _Float16* Wt = (z == 0) ? WtQ : (z == 1) ? WtK : WtV;

    v8f acc[2][4];
#pragma unroll
    for (int mt = 0; mt < 2; ++mt)
#pragma unroll
        for (int j = 0; j < 4; ++j) acc[mt][j] = zero8();

    for (int k0 = 0; k0 < D; k0 += 32) {
        // A frags: lane row = mbase + mt*16 + (lane%16); K slots v0..3 = half*8+0..7, v4..7 = 16+half*8+0..7
        const _Float16* ar0 = X + (size_t)(mbase + n) * D + k0;
        const _Float16* ar1 = X + (size_t)(mbase + 16 + n) * D + k0;
        v16h a0 = cat16(ld8(ar0 + half * 8), ld8(ar0 + 16 + half * 8));
        v16h a1 = cat16(ld8(ar1 + half * 8), ld8(ar1 + 16 + half * 8));
        __builtin_prefetch(ar0 + 64, 0, 3);   // global_prefetch_b8, two k-steps ahead
        __builtin_prefetch(ar1 + 64, 0, 3);
#pragma unroll
        for (int s4 = 0; s4 < 4; ++s4) {
            // B frag: lane col = nbase+s4*16+(lane%16); K slots v0..3 = half*8+0..7, v4..7 = 16+half*8+0..7
            const _Float16* br = Wt + (size_t)(nbase + s4 * 16 + n) * D + k0 + half * 8;
            v16h bfr = cat16(ld8(br), ld8(br + 16));
            __builtin_prefetch(br + 64, 0, 3);
            acc[0][s4] = wmma16(a0, bfr, acc[0][s4]);
            acc[1][s4] = wmma16(a1, bfr, acc[1][s4]);
        }
    }

    const float* bias = (z == 0) ? bq : (z == 1) ? bk : bv;
#pragma unroll
    for (int s4 = 0; s4 < 4; ++s4) {
        int ng = nbase + s4 * 16 + n;            // global col
        int hh = ng >> 6, dd = ng & 63;          // head, dim (nbase is 64-aligned)
        float bval = bias[ng];
#pragma unroll
        for (int mt = 0; mt < 2; ++mt) {
#pragma unroll
            for (int r = 0; r < 8; ++r) {
                int mg  = mbase + mt * 16 + half * 8 + r;  // global row (C layout)
                int bi  = mg >> 11, ss = mg & 2047;        // batch, seq
                float v = acc[mt][s4][r] + bval;
                size_t bh = (size_t)(bi * H + hh);
                if (z == 0)      Q[(bh * S + ss) * HD + dd] = (_Float16)(v * 0.125f); // fold 1/sqrt(64)
                else if (z == 1) K[(bh * S + ss) * HD + dd] = (_Float16)v;
                else             Vt[(bh * HD + dd) * S + ss] = (_Float16)v;           // transposed V
            }
        }
    }
}

// ---------- flash attention (transposed-score formulation) ----------
// grid (S/16, H, B); block 32. One wave handles a 16-query tile, streams keys in tiles of 32.
__global__ void k_attn(const _Float16* __restrict__ Q,
                       const _Float16* __restrict__ K,
                       const _Float16* __restrict__ Vt,
                       const float* __restrict__ mask,   // [B,1,1,S]
                       _Float16* __restrict__ Ctx) {     // f16 [M][D]
    const int lane = threadIdx.x & 31;
    const int n = lane & 15, half = lane >> 4;           // n = query within tile
    const int qbase = blockIdx.x * 16;
    const int h = blockIdx.y, b = blockIdx.z;
    const size_t bh = (size_t)(b * H + h);
    const _Float16* Qh = Q  + bh * (size_t)S * HD;
    const _Float16* Kh = K  + bh * (size_t)S * HD;
    const _Float16* Vh = Vt + bh * (size_t)HD * S;
    const float*    mb = mask + (size_t)b * S;

    // B frags of Q^T: lane col = query, K slots = head-dim (contiguous in Q row)
    const _Float16* qr = Qh + (size_t)(qbase + n) * HD + half * 8;
    const v16h bq0 = cat16(ld8(qr),      ld8(qr + 16));   // d = 0..31
    const v16h bq1 = cat16(ld8(qr + 32), ld8(qr + 48));   // d = 32..63

    float mrun = -INFINITY, lrun = 0.f;
    v8f acc[4];                                           // ctx^T: 4 chunks of (16 d x 16 q)
#pragma unroll
    for (int j = 0; j < 4; ++j) acc[j] = zero8();

    for (int kt = 0; kt < S; kt += 32) {
        // A frags of K tile: lane row = key, K slots = head-dim
        const _Float16* kr0 = Kh + (size_t)(kt + n) * HD;
        const _Float16* kr1 = Kh + (size_t)(kt + 16 + n) * HD;
        v16h a00 = cat16(ld8(kr0 + half * 8),      ld8(kr0 + 16 + half * 8));
        v16h a01 = cat16(ld8(kr0 + 32 + half * 8), ld8(kr0 + 48 + half * 8));
        v16h a10 = cat16(ld8(kr1 + half * 8),      ld8(kr1 + 16 + half * 8));
        v16h a11 = cat16(ld8(kr1 + 32 + half * 8), ld8(kr1 + 48 + half * 8));

        v8f c0 = zero8(), c1 = zero8();                   // S^T tiles: rows=keys, cols=queries
        c0 = wmma16(a00, bq0, c0); c0 = wmma16(a01, bq1, c0);
        c1 = wmma16(a10, bq0, c1); c1 = wmma16(a11, bq1, c1);

        // per-lane: this lane's 16 keys for its query column
        float p0[8], p1[8], lmax = -INFINITY;
#pragma unroll
        for (int r = 0; r < 8; ++r) {
            float s0 = c0[r] + mb[kt +      half * 8 + r];
            float s1 = c1[r] + mb[kt + 16 + half * 8 + r];
            p0[r] = s0; p1[r] = s1;
            lmax = fmaxf(lmax, fmaxf(s0, s1));
        }
        float tmax  = fmaxf(lmax, xor16f(lmax));          // full per-query max (both halves agree)
        float mnew  = fmaxf(mrun, tmax);
        float alpha = __expf(mrun - mnew);

        float ps = 0.f;
#pragma unroll
        for (int r = 0; r < 8; ++r) {
            p0[r] = __expf(p0[r] - mnew);
            p1[r] = __expf(p1[r] - mnew);
            ps += p0[r] + p1[r];
        }
        lrun = lrun * alpha + ps;                         // per-lane partial (half-split over keys)
        mrun = mnew;
#pragma unroll
        for (int j = 0; j < 4; ++j)
#pragma unroll
            for (int r = 0; r < 8; ++r) acc[j][r] *= alpha;

        // Repack P into B-frag layout: lane col = query, K slot j -> key kt + half*16 + j
        v16h bp;
#pragma unroll
        for (int r = 0; r < 8; ++r) {
            float o0 = xor16f(p0[r]);                     // partner's keys
            float o1 = xor16f(p1[r]);
            float lo = half ? o1 : p0[r];                 // K slots 0..7
            float hi = half ? p1[r] : o0;                 // K slots 8..15
            bp[r]     = (_Float16)lo;
            bp[r + 8] = (_Float16)hi;
        }

        // ctx^T += Vt_chunk(16d x 32keys) * P^T(32keys x 16q)
#pragma unroll
        for (int j = 0; j < 4; ++j) {
            const _Float16* vr = Vh + (size_t)(j * 16 + n) * S + kt;
            v16h av = cat16(ld8(vr + half * 16), ld8(vr + half * 16 + 8));
            acc[j] = wmma16(av, bp, acc[j]);
        }
    }

    float ltot = lrun + xor16f(lrun);
    float inv  = 1.f / ltot;
#pragma unroll
    for (int j = 0; j < 4; ++j) {
#pragma unroll
        for (int r = 0; r < 8; ++r) {
            int dd = j * 16 + half * 8 + r;               // C layout row = d
            size_t row = (size_t)b * S + qbase + n;       // col = query
            Ctx[row * D + h * HD + dd] = (_Float16)(acc[j][r] * inv);
        }
    }
}

// ---------- output projection + bias + residual ----------
// grid (M/32, D/64); block 32; 32x64 strip per wave like the QKV GEMM
__global__ void k_proj(const _Float16* __restrict__ Ctx,
                       const _Float16* __restrict__ WtO,
                       const float* __restrict__ bo,
                       const float* __restrict__ hid,
                       float* __restrict__ xout) {
    const int lane = threadIdx.x & 31;
    const int n = lane & 15, half = lane >> 4;
    const int mbase = blockIdx.x * 32;
    const int nbase = blockIdx.y * 64;

    v8f acc[2][4];
#pragma unroll
    for (int mt = 0; mt < 2; ++mt)
#pragma unroll
        for (int j = 0; j < 4; ++j) acc[mt][j] = zero8();

    for (int k0 = 0; k0 < D; k0 += 32) {
        const _Float16* ar0 = Ctx + (size_t)(mbase + n) * D + k0;
        const _Float16* ar1 = Ctx + (size_t)(mbase + 16 + n) * D + k0;
        v16h a0 = cat16(ld8(ar0 + half * 8), ld8(ar0 + 16 + half * 8));
        v16h a1 = cat16(ld8(ar1 + half * 8), ld8(ar1 + 16 + half * 8));
        __builtin_prefetch(ar0 + 64, 0, 3);
        __builtin_prefetch(ar1 + 64, 0, 3);
#pragma unroll
        for (int s4 = 0; s4 < 4; ++s4) {
            const _Float16* br = WtO + (size_t)(nbase + s4 * 16 + n) * D + k0 + half * 8;
            v16h bfr = cat16(ld8(br), ld8(br + 16));
            __builtin_prefetch(br + 64, 0, 3);
            acc[0][s4] = wmma16(a0, bfr, acc[0][s4]);
            acc[1][s4] = wmma16(a1, bfr, acc[1][s4]);
        }
    }
#pragma unroll
    for (int s4 = 0; s4 < 4; ++s4) {
        int ng = nbase + s4 * 16 + n;
        float bval = bo[ng];
#pragma unroll
        for (int mt = 0; mt < 2; ++mt) {
#pragma unroll
            for (int r = 0; r < 8; ++r) {
                int mg = mbase + mt * 16 + half * 8 + r;
                xout[(size_t)mg * D + ng] = acc[mt][s4][r] + bval + hid[(size_t)mg * D + ng];
            }
        }
    }
}

// ---------- LayerNorm ----------
__global__ void k_ln(const float* __restrict__ x, const float* __restrict__ g,
                     const float* __restrict__ be, float* __restrict__ out) {
    __shared__ float s1[256], s2[256];
    const int row = blockIdx.x, t = threadIdx.x;
    const float* xr = x + (size_t)row * D;
    float a = 0.f, b = 0.f;
    for (int i = t; i < D; i += 256) { float v = xr[i]; a += v; b += v * v; }
    s1[t] = a; s2[t] = b; __syncthreads();
    for (int off = 128; off > 0; off >>= 1) {
        if (t < off) { s1[t] += s1[t + off]; s2[t] += s2[t + off]; }
        __syncthreads();
    }
    float mu  = s1[0] * (1.f / D);
    float var = s2[0] * (1.f / D) - mu * mu;
    float inv = rsqrtf(var + 1e-12f);
    for (int i = t; i < D; i += 256)
        out[(size_t)row * D + i] = (xr[i] - mu) * inv * g[i] + be[i];
}

// ---------- host ----------
extern "C" void kernel_launch(void* const* d_in, const int* in_sizes, int n_in,
                              void* d_out, int out_size, void* d_ws, size_t ws_size,
                              hipStream_t stream) {
    (void)in_sizes; (void)n_in; (void)out_size; (void)ws_size;
    const float* hid   = (const float*)d_in[0];
    const float* mask  = (const float*)d_in[1];
    const float* Wq    = (const float*)d_in[2];
    const float* bq    = (const float*)d_in[3];
    const float* Wk    = (const float*)d_in[4];
    const float* bk    = (const float*)d_in[5];
    const float* Wv    = (const float*)d_in[6];
    const float* bv    = (const float*)d_in[7];
    const float* Wo    = (const float*)d_in[8];
    const float* bo    = (const float*)d_in[9];
    const float* gamma = (const float*)d_in[10];
    const float* beta  = (const float*)d_in[11];

    char* ws = (char*)d_ws;
    const size_t MB = 1024 * 1024;
    _Float16* xf16 = (_Float16*)(ws + 0);            //  8 MB: X f16 [4096][1024]
    _Float16* wtq  = (_Float16*)(ws + 8  * MB);      //  2 MB each: Wt f16 [N][K]
    _Float16* wtk  = (_Float16*)(ws + 10 * MB);
    _Float16* wtv  = (_Float16*)(ws + 12 * MB);
    _Float16* wto  = (_Float16*)(ws + 14 * MB);
    _Float16* qbuf = (_Float16*)(ws + 16 * MB);      //  8 MB: Q f16 [B*H][S][HD] (pre-scaled)
    _Float16* kbuf = (_Float16*)(ws + 24 * MB);      //  8 MB
    _Float16* vtb  = (_Float16*)(ws + 32 * MB);      //  8 MB: V^T f16 [B*H][HD][S]
    _Float16* ctx  = (_Float16*)(ws + 40 * MB);      //  8 MB: ctx f16 [4096][1024]
    float*    xbuf = (float*)   (ws + 48 * MB);      // 16 MB: proj+residual f32

    k_cvt_f16<<<(M * D + 255) / 256, 256, 0, stream>>>(hid, xf16, M * D);
    k_transpose_f16<<<(D * D) / 256, 256, 0, stream>>>(Wq, wtq);
    k_transpose_f16<<<(D * D) / 256, 256, 0, stream>>>(Wk, wtk);
    k_transpose_f16<<<(D * D) / 256, 256, 0, stream>>>(Wv, wtv);
    k_transpose_f16<<<(D * D) / 256, 256, 0, stream>>>(Wo, wto);

    k_qkv_gemm<<<dim3(M / 32, D / 64, 3), 32, 0, stream>>>(
        xf16, wtq, wtk, wtv, bq, bk, bv, qbuf, kbuf, vtb);

    k_attn<<<dim3(S / 16, H, Bb), 32, 0, stream>>>(qbuf, kbuf, vtb, mask, ctx);

    k_proj<<<dim3(M / 32, D / 64), 32, 0, stream>>>(ctx, wto, bo, hid, xbuf);

    k_ln<<<M, 256, 0, stream>>>(xbuf, gamma, beta, (float*)d_out);
}
